// DifferentialAttention_28802050687601
// MI455X (gfx1250) — matrix operand order repaired and checked by reading, not compiled, not run on hardware
//
#include <hip/hip_runtime.h>
#include <hip/hip_bf16.h>

// ---------------------------------------------------------------------------
// DifferentialAttention on MI455X (gfx1250, wave32, WMMA 16x16x32 f16,
// async global->LDS staging via ASYNCcnt)
// ---------------------------------------------------------------------------
constexpr int cB   = 4;
constexpr int cN   = 2304;   // 48*48 grid
constexpr int cC   = 512;
constexpr int cH   = 8;
constexpr int cHKV = 4;
constexpr int cD   = 64;     // head dim
constexpr int cSIDE = 48;
constexpr int cWIN  = 12;
constexpr int cM   = cB * cN;  // 9216 rows for all projection GEMMs

typedef __attribute__((ext_vector_type(16))) _Float16 v16h;
typedef __attribute__((ext_vector_type(8)))  _Float16 v8h;
typedef __attribute__((ext_vector_type(8)))  float    v8f;

// D = A(16x32 f16) x B(32x16 f16) + C(16x16 f32)
__device__ __forceinline__ v8f wmma_16x16x32(v16h a, v16h b, v8f c) {
    return __builtin_amdgcn_wmma_f32_16x16x32_f16(
        /*neg_a=*/false, a, /*neg_b=*/false, b,
        /*c_mod=*/(short)0, c, /*reuse_a=*/false, /*reuse_b=*/false);
}

// Async global->LDS copy, 16B per lane (CDNA5 ISA: GLOBAL_LOAD_ASYNC_TO_LDS_B128).
// lds_off = low 32 bits of the generic shared pointer (== wave LDS offset).
__device__ __forceinline__ void async_b128(unsigned lds_off, const void* gaddr) {
    asm volatile("global_load_async_to_lds_b128 %0, %1, off"
                 :: "v"(lds_off), "v"((unsigned long long)(uintptr_t)gaddr)
                 : "memory");
}
__device__ __forceinline__ void wait_async0() {
    asm volatile("s_wait_asynccnt 0x0" ::: "memory");
}
__device__ __forceinline__ unsigned lds_off32(const void* p) {
    return (unsigned)(uintptr_t)p;   // generic LDS addr: [31:0] = LDS offset
}

// A-fragment (16x32, row-major in LDS, row stride ld halves).
// contiguous k order (same as B loaders): lane L holds row m=L&15; VGPR v halves (2v,2v+1) are
// K = (L>>4)*16 + v*2 (+0/+1).
__device__ __forceinline__ v16h load_frag_a(const _Float16* p, int ld) {
    int lane = threadIdx.x & 31;
    int m = lane & 15, hi = lane >> 4;
    v16h a;
#pragma unroll
    for (int v = 0; v < 8; ++v) {
        int k = hi * 16 + v * 2;
        a[2 * v]     = p[m * ld + k];
        a[2 * v + 1] = p[m * ld + k + 1];
    }
    return a;
}

// B-fragment (32x16) from a transposed tile Bt[n][k] (row stride ld halves):
// lane L holds column n=L&15; element e is K = (L>>4)*16 + e.
__device__ __forceinline__ v16h load_frag_b(const _Float16* bt, int ld) {
    int lane = threadIdx.x & 31;
    int n = lane & 15, hi = lane >> 4;
    const _Float16* p = bt + n * ld + hi * 16;
    v16h b;
#pragma unroll
    for (int e = 0; e < 16; ++e) b[e] = p[e];
    return b;
}

// B-fragment from a row-major B[k][n] tile (row stride ldn halves).
__device__ __forceinline__ v16h load_frag_b_rm(const _Float16* bm, int ldn, int ncol) {
    int lane = threadIdx.x & 31;
    int n = (lane & 15) + ncol, hi = lane >> 4;
    v16h b;
#pragma unroll
    for (int e = 0; e < 16; ++e) b[e] = bm[(hi * 16 + e) * ldn + n];
    return b;
}

// ---------------------------------------------------------------------------
__global__ __launch_bounds__(256)
void cvt_f32_f16(const float* __restrict__ in, _Float16* __restrict__ out, int n) {
    int i = blockIdx.x * 256 + threadIdx.x;
    if (i < n) out[i] = (_Float16)in[i];
}

// ---------------------------------------------------------------------------
// Generic f16 WMMA GEMM: C[M,Nn] = A[M,K] * B[K,Nn] (+bias), f32 out.
// Tile 128x64, K-step 32, 256 threads = 8 waves, wave w -> 16-row strip.
// A tile staged with async global->LDS; B tile transposed via LDS scatter.
// ---------------------------------------------------------------------------
__global__ __launch_bounds__(256)
void gemm_f16f32(const _Float16* __restrict__ A, const _Float16* __restrict__ Bw,
                 const float* __restrict__ bias, float* __restrict__ Cout,
                 int M, int Nn, int K) {
    __shared__ _Float16 sA[128 * 32];
    __shared__ _Float16 sBt[64 * 32];   // B tile transposed: [n][k]
    const int wave = threadIdx.x >> 5;
    const int tiles_n = Nn / 64;
    const int row0 = (blockIdx.x / tiles_n) * 128;
    const int col0 = (blockIdx.x % tiles_n) * 64;
    v8f acc[4] = {};

    const int ar = threadIdx.x >> 1;          // A stage: row 0..127
    const int ac = (threadIdx.x & 1) * 16;    // A stage: col 0/16
    const int bk = threadIdx.x >> 3;          // B stage: k 0..31
    const int bn = (threadIdx.x & 7) * 8;     // B stage: n 0..56
    const unsigned aoff = lds_off32(sA + ar * 32 + ac);

    for (int k0 = 0; k0 < K; k0 += 32) {
        {   // stage A asynchronously: 128x32 halves, 2 x b128 per thread
            const _Float16* g = A + (size_t)(row0 + ar) * K + (k0 + ac);
            async_b128(aoff, g);
            async_b128(aoff + 16, g + 8);
        }
        {   // stage Bt: read 8 contiguous cols, scatter transposed to LDS
            v8h bv = *(const v8h*)(Bw + (size_t)(k0 + bk) * Nn + col0 + bn);
#pragma unroll
            for (int i = 0; i < 8; ++i) sBt[(bn + i) * 32 + bk] = bv[i];
        }
        if (k0 + 32 < K) {                    // prefetch next K slice
            __builtin_prefetch(A + (size_t)(row0 + ar) * K + (k0 + 32 + ac), 0, 1);
            __builtin_prefetch(Bw + (size_t)(k0 + 32 + bk) * Nn + col0 + bn, 0, 1);
        }
        wait_async0();
        __syncthreads();
        v16h a = load_frag_a(sA + (wave * 16) * 32, 32);
#pragma unroll
        for (int nn = 0; nn < 4; ++nn) {
            v16h b = load_frag_b(sBt + (nn * 16) * 32, 32);
            acc[nn] = wmma_16x16x32(a, b, acc[nn]);
        }
        __syncthreads();
    }
    // C layout: VGPR j -> row j + 8*(lane>>4), col lane&15
    int lane = threadIdx.x & 31;
    int nloc = lane & 15, hi = lane >> 4;
#pragma unroll
    for (int nn = 0; nn < 4; ++nn) {
        int ccol = col0 + nn * 16 + nloc;
        float bv = bias ? bias[ccol] : 0.0f;   // hoisted: column-only dependence
#pragma unroll
        for (int j = 0; j < 8; ++j) {
            int r = row0 + wave * 16 + j + 8 * hi;
            Cout[(size_t)r * Nn + ccol] = acc[nn][j] + bv;
        }
    }
}

// ---------------------------------------------------------------------------
// L2-norm + 2D axial RoPE + head-major f16 repack.
// One wave per (b, head, token); d=64 -> 2 elements/lane, rotate-half pair
// (j, j+32) lives inside one lane.
// ---------------------------------------------------------------------------
__global__ __launch_bounds__(256)
void norm_rope_pack(const float* __restrict__ in, _Float16* __restrict__ out,
                    int nheads, int in_stride, int in_off, int do_rope) {
    int wid  = blockIdx.x * 8 + (threadIdx.x >> 5);
    int lane = threadIdx.x & 31;
    int total = cB * nheads * cN;
    if (wid >= total) return;
    int n = wid % cN;
    int h = (wid / cN) % nheads;
    int b = wid / (cN * nheads);
    const float* src = in + ((size_t)b * cN + n) * in_stride + in_off + h * cD;
    float a = src[lane];
    float c = src[lane + 32];
    if (do_rope) {
        float ss = a * a + c * c;
#pragma unroll
        for (int m = 16; m >= 1; m >>= 1) ss += __shfl_xor(ss, m, 32);
        float inv = rsqrtf(ss + 1e-12f);
        a *= inv; c *= inv;
        // theta[j]: j%32<16 -> row-pos * base^{-(j%16)/16}; else col-pos.
        float pos  = (lane < 16) ? (float)(n / cSIDE) : (float)(n % cSIDE);
        float freq = __expf(-9.210340372f * (float)(lane & 15) / 16.0f); // 10000^-i/16
        float th = pos * freq;
        float cs = __cosf(th), sn = __sinf(th);
        float ra = a * cs - c * sn;   // out[j]    = x[j]*cos - x[j+32]*sin
        float rc = c * cs + a * sn;   // out[j+32] = x[j+32]*cos + x[j]*sin
        a = ra; c = rc;
    }
    _Float16* dst = out + (((size_t)b * nheads + h) * cN + n) * cD;
    dst[lane]      = (_Float16)a;
    dst[lane + 32] = (_Float16)c;
}

// ---------------------------------------------------------------------------
// Sliding-window attention, one wave per (b, h, 16-query tile).
// Key tiles of 32 (one grid row slice): S = Q*K^T via 4 WMMAs (K-dim 64),
// online softmax, O += P*V via 4 WMMAs (K-dim 32 = 32 keys).
// K/V tiles staged with async global->LDS (ASYNCcnt).
// ---------------------------------------------------------------------------
__global__ __launch_bounds__(32)
void attn_window(const _Float16* __restrict__ Qh, const _Float16* __restrict__ Kh,
                 const _Float16* __restrict__ Vh, float* __restrict__ Obuf) {
    __shared__ _Float16 sQ[16 * 64];
    __shared__ _Float16 sK[32 * 64];
    __shared__ _Float16 sV[32 * 64];
    __shared__ _Float16 sP[16 * 32];
    const int NT = cN / 16;                    // 144 query tiles
    int qt = blockIdx.x % NT;
    int h  = (blockIdx.x / NT) % cH;
    int b  = blockIdx.x / (NT * cH);
    int lane = threadIdx.x;
    int hkv = h >> 1;                          // GQA: rep = H/HKV = 2
    const _Float16* qsrc = Qh + (((size_t)b * cH + h) * cN + qt * 16) * cD;
    const _Float16* ksrc = Kh + ((size_t)b * cHKV + hkv) * cN * cD;
    const _Float16* vsrc = Vh + ((size_t)b * cHKV + hkv) * cN * cD;

#pragma unroll
    for (int i = 0; i < 4; ++i) {              // stage 16x64 Q tile
        int c = (lane + i * 32) * 8;
        *(v8h*)(sQ + c) = *(const v8h*)(qsrc + c);
    }
    __syncthreads();
    v16h a0 = load_frag_a(sQ, 64);             // k 0..31
    v16h a1 = load_frag_a(sQ + 32, 64);        // k 32..63

    const int q0 = qt * 16;
    const int qrow = q0 / cSIDE, qcol0 = q0 % cSIDE;  // 16 queries share a grid row
    float mrun[8], lrun[8];
    v8f O[4] = {};
#pragma unroll
    for (int j = 0; j < 8; ++j) { mrun[j] = -1e30f; lrun[j] = 0.f; }
    const int nloc = lane & 15, hi = lane >> 4;
    const unsigned koff = lds_off32(sK + lane * 64);
    const unsigned voff = lds_off32(sV + lane * 64);

    int krlo = qrow - cWIN; if (krlo < 0) krlo = 0;
    int krhi = qrow + cWIN; if (krhi > cSIDE - 1) krhi = cSIDE - 1;
    int cstart = qcol0 - cWIN; if (cstart < 0) cstart = 0;
    int cend = qcol0 + 16 + cWIN; if (cend > cSIDE) cend = cSIDE;

    for (int kr = krlo; kr <= krhi; ++kr) {
        for (int kc0 = cstart; kc0 < cend; kc0 += 32) {
            __syncthreads();
            int kcol = kc0 + lane;
            int ktok = kr * cSIDE + (kcol < cSIDE ? kcol : cSIDE - 1);
            const _Float16* kg = ksrc + (size_t)ktok * cD;
            const _Float16* vg = vsrc + (size_t)ktok * cD;
#pragma unroll
            for (int i = 0; i < 8; ++i) {      // async-stage one key row per lane
                async_b128(koff + i * 16, kg + i * 8);
                async_b128(voff + i * 16, vg + i * 8);
            }
            wait_async0();
            __syncthreads();

            v8f S0 = {}, S1 = {};
            {
                v16h b00 = load_frag_b(sK, 64);            // keys 0-15, k 0-31
                v16h b01 = load_frag_b(sK + 32, 64);       // keys 0-15, k 32-63
                S0 = wmma_16x16x32(a0, b00, S0);
                S0 = wmma_16x16x32(a1, b01, S0);
                v16h b10 = load_frag_b(sK + 16 * 64, 64);  // keys 16-31
                v16h b11 = load_frag_b(sK + 16 * 64 + 32, 64);
                S1 = wmma_16x16x32(a0, b10, S1);
                S1 = wmma_16x16x32(a1, b11, S1);
            }
            // mask + online softmax (row r = j + 8*hi across 16 lanes)
#pragma unroll
            for (int j = 0; j < 8; ++j) {
                int m = j + 8 * hi;
                int qc = qcol0 + m;
                int ka = kc0 + nloc, kb = ka + 16;
                float s0 = S0[j] * 0.125f, s1 = S1[j] * 0.125f;
                int dr = kr - qrow; dr = dr < 0 ? -dr : dr;
                int da = ka - qc;   da = da < 0 ? -da : da;
                int db = kb - qc;   db = db < 0 ? -db : db;
                if (!(ka < cSIDE && dr <= cWIN && da <= cWIN)) s0 = -1e30f;
                if (!(kb < cSIDE && dr <= cWIN && db <= cWIN)) s1 = -1e30f;
                float mx = s0 > s1 ? s0 : s1;
#pragma unroll
                for (int d = 1; d <= 8; d <<= 1) {
                    float o = __shfl_xor(mx, d, 32);
                    mx = mx > o ? mx : o;
                }
                float newm = mrun[j] > mx ? mrun[j] : mx;
                float sc = __expf(mrun[j] - newm);
                float p0 = __expf(s0 - newm), p1 = __expf(s1 - newm);
                float rs = p0 + p1;
#pragma unroll
                for (int d = 1; d <= 8; d <<= 1) rs += __shfl_xor(rs, d, 32);
                lrun[j] = lrun[j] * sc + rs;
                mrun[j] = newm;
#pragma unroll
                for (int t = 0; t < 4; ++t) O[t][j] *= sc;
                sP[m * 32 + nloc]      = (_Float16)p0;
                sP[m * 32 + nloc + 16] = (_Float16)p1;
            }
            __syncthreads();
            v16h pa = load_frag_a(sP, 32);     // P as 16x32 A matrix
#pragma unroll
            for (int t = 0; t < 4; ++t) {
                v16h bv = load_frag_b_rm(sV, 64, t * 16);  // V rows = K-dim
                O[t] = wmma_16x16x32(pa, bv, O[t]);
            }
        }
    }
    float* obase = Obuf + (((size_t)b * cH + h) * cN + q0) * cD;
#pragma unroll
    for (int t = 0; t < 4; ++t)
#pragma unroll
        for (int j = 0; j < 8; ++j) {
            int m = j + 8 * hi;
            obase[(size_t)m * cD + t * 16 + nloc] = O[t][j] / lrun[j];
        }
}

// ---------------------------------------------------------------------------
// out[b,n,h*64+j] = O1 - sigmoid(lambda_p[h]) * O2, f16 token-major
// ---------------------------------------------------------------------------
__global__ __launch_bounds__(256)
void combine_heads(const float* __restrict__ O1, const float* __restrict__ O2,
                   const float* __restrict__ lamp, _Float16* __restrict__ out) {
    int i = blockIdx.x * 256 + threadIdx.x;
    if (i >= cB * cN * cC) return;
    int c = i % cC;
    int n = (i / cC) % cN;
    int b = i / (cC * cN);
    int h = c >> 6, j = c & 63;
    float lam = 1.f / (1.f + __expf(-lamp[h]));
    size_t s = (((size_t)b * cH + h) * cN + n) * cD + j;
    out[i] = (_Float16)(O1[s] - lam * O2[s]);
}

// ---------------------------------------------------------------------------
extern "C" void kernel_launch(void* const* d_in, const int* in_sizes, int n_in,
                              void* d_out, int out_size, void* d_ws, size_t ws_size,
                              hipStream_t stream) {
    (void)in_sizes; (void)n_in; (void)out_size; (void)ws_size;
    const float* x    = (const float*)d_in[0];
    const float* Wq   = (const float*)d_in[1];
    const float* Wk   = (const float*)d_in[2];
    const float* Wv1  = (const float*)d_in[3];
    const float* Wv2  = (const float*)d_in[4];
    const float* lamp = (const float*)d_in[5];
    const float* Wp   = (const float*)d_in[6];
    const float* bp   = (const float*)d_in[7];
    float* out = (float*)d_out;

    char* ws = (char*)d_ws;
    size_t off = 0;
    auto take = [&](size_t bytes) -> char* {
        char* p = ws + off;
        off += (bytes + 255) & ~(size_t)255;
        return p;
    };
    _Float16* xh   = (_Float16*)take((size_t)cM * 512 * 2);
    _Float16* Wqh  = (_Float16*)take((size_t)512 * 1024 * 2);
    _Float16* Wkh  = (_Float16*)take((size_t)512 * 512 * 2);
    _Float16* Wv1h = (_Float16*)take((size_t)512 * 256 * 2);
    _Float16* Wv2h = (_Float16*)take((size_t)512 * 256 * 2);
    _Float16* Wph  = (_Float16*)take((size_t)512 * 512 * 2);
    float* qf  = (float*)take((size_t)cM * 1024 * 4);
    float* kf  = (float*)take((size_t)cM * 512 * 4);
    float* v1f = (float*)take((size_t)cM * 256 * 4);
    float* v2f = (float*)take((size_t)cM * 256 * 4);
    const size_t hd  = (size_t)cB * cH * cN * cD;    // 4.7M
    const size_t hdk = (size_t)cB * cHKV * cN * cD;  // 2.36M
    _Float16* q1h = (_Float16*)take(hd * 2);
    _Float16* q2h = (_Float16*)take(hd * 2);
    _Float16* k1h = (_Float16*)take(hdk * 2);
    _Float16* k2h = (_Float16*)take(hdk * 2);
    _Float16* v1h = (_Float16*)take(hdk * 2);
    _Float16* v2h = (_Float16*)take(hdk * 2);
    // reuse dead f32 buffers: qf (37.7MB) holds O1+O2, kf holds combined f16
    float* O1 = qf;
    float* O2 = qf + hd;
    _Float16* attnh = (_Float16*)kf;

    auto cvt = [&](const float* s, _Float16* d, size_t n) {
        cvt_f32_f16<<<dim3((unsigned)((n + 255) / 256)), dim3(256), 0, stream>>>(s, d, (int)n);
    };
    cvt(x,   xh,   (size_t)cM * 512);
    cvt(Wq,  Wqh,  (size_t)512 * 1024);
    cvt(Wk,  Wkh,  (size_t)512 * 512);
    cvt(Wv1, Wv1h, (size_t)512 * 256);
    cvt(Wv2, Wv2h, (size_t)512 * 256);
    cvt(Wp,  Wph,  (size_t)512 * 512);

    auto gemm = [&](const _Float16* A, const _Float16* Bm, const float* bias,
                    float* C, int Nn) {
        gemm_f16f32<<<dim3((unsigned)((cM / 128) * (Nn / 64))), dim3(256), 0, stream>>>(
            A, Bm, bias, C, cM, Nn, 512);
    };
    gemm(xh, Wqh,  nullptr, qf,  1024);
    gemm(xh, Wkh,  nullptr, kf,  512);
    gemm(xh, Wv1h, nullptr, v1f, 256);
    gemm(xh, Wv2h, nullptr, v2f, 256);

    auto rope = [&](const float* in, _Float16* o, int nh, int stride, int ioff, int dorope) {
        int waves = cB * nh * cN;
        norm_rope_pack<<<dim3((unsigned)((waves + 7) / 8)), dim3(256), 0, stream>>>(
            in, o, nh, stride, ioff, dorope);
    };
    rope(qf,  q1h, cH,   1024, 0,   1);
    rope(qf,  q2h, cH,   1024, 512, 1);
    rope(kf,  k1h, cHKV, 512,  0,   1);
    rope(kf,  k2h, cHKV, 512,  256, 1);
    rope(v1f, v1h, cHKV, 256,  0,   0);
    rope(v2f, v2h, cHKV, 256,  0,   0);

    dim3 agrid((unsigned)(cB * cH * (cN / 16)));
    attn_window<<<agrid, dim3(32), 0, stream>>>(q1h, k1h, v1h, O1);
    attn_window<<<agrid, dim3(32), 0, stream>>>(q2h, k2h, v2h, O2);

    combine_heads<<<dim3((unsigned)((cB * cN * cC + 255) / 256)), dim3(256), 0, stream>>>(
        O1, O2, lamp, attnh);

    gemm(attnh, Wph, bp, out, 512);
}
